// MSAAttention_31473520345695
// MI455X (gfx1250) — matrix operand order repaired and checked by reading, not compiled, not run on hardware
//
#include <hip/hip_runtime.h>
#include <hip/hip_bf16.h>
#include <hip/hip_fp16.h>

#define BB 4
#define SS 1024
#define DD 512
#define HH 8
#define DH 64

typedef __attribute__((ext_vector_type(16))) _Float16 v16h;
typedef __attribute__((ext_vector_type(8)))  _Float16 v8h;
typedef __attribute__((ext_vector_type(8)))  float    v8f;

// softmax(q.k / sqrt(64)) == exp2( (q * log2e/8) . k ) / rowsum
// -> fold both the 1/sqrt(dh) scale and the exp->exp2 conversion into q once.
#define QSCALE 0.18033688011112042f   // log2(e) / 8

// ---------------------------------------------------------------------------
// Kernel 1: per-head linear projections q = (xWq+bq)*QSCALE, k = xWk+bk.
// Writes f16 into workspace, layout [B,H,S,DH] so each (b,h) is a contiguous
// [S,64] matrix with 128B-aligned rows (ideal for WMMA fragment loads).
// Consecutive threads vary the output feature e -> W reads are coalesced,
// x row is broadcast within the wave. ~0.27 GFLOP total: VALU is plenty.
// ---------------------------------------------------------------------------
__global__ void msa_proj_kernel(const float* __restrict__ x,
                                const float* __restrict__ Wq,
                                const float* __restrict__ bq,
                                const float* __restrict__ Wk,
                                const float* __restrict__ bk,
                                _Float16* __restrict__ qws,
                                _Float16* __restrict__ kws) {
    const int NQ = BB * HH * SS * DH;                       // 2^21
    int t = blockIdx.x * blockDim.x + threadIdx.x;          // [0, 2*NQ)
    int which = t >= NQ;                                    // 0 = q, 1 = k
    int r = t - which * NQ;                                 // ((b*H+h)*S+s)*64+e
    int e = r & (DH - 1);
    int s = (r >> 6) & (SS - 1);
    int h = (r >> 16) & (HH - 1);
    int b = r >> 19;

    const float* W    = which ? Wk : Wq;
    const float* bias = which ? bk : bq;
    _Float16*    o    = which ? kws : qws;
    const float  scl  = which ? 1.0f : QSCALE;

    const float* xr   = x + ((size_t)(b * SS + s)) * DD + h * DH;   // head slice
    const float* wcol = W + ((size_t)h) * DH * DH + e;              // W[h, d, e]
    float acc = bias[h * DH + e];
#pragma unroll 8
    for (int d = 0; d < DH; ++d) acc += xr[d] * wcol[d * DH];
    o[r] = (_Float16)(acc * scl);
}

// ---------------------------------------------------------------------------
// Kernel 2: one wave (32 threads) per 16-row score block of one (b,h).
//   - q tile A-fragments held in VGPRs (ISA 16-bit A layout:
//     lane<16 -> K {0..7}U{16..23}, lane>=16 -> K {8..15}U{24..31})
//   - per 16-key tile: B-fragment = one key row/lane, same K split as A
//     (ISA 16-bit B layout), 2x v_wmma_f32_16x16x32_f16 (K=64)
//   - raw v_exp_f32 applied in-register (scale pre-folded into q; scores are
//     O(1), so no subnormal/overflow range fixup needed), stored to the LDS
//     score panel; row-sums accumulated in registers (max-free softmax)
//   - half-wave shfl_xor reduction for row sums, then coalesced float4 store
// ---------------------------------------------------------------------------
__global__ void msa_attn_kernel(const _Float16* __restrict__ qws,
                                const _Float16* __restrict__ kws,
                                float* __restrict__ out) {
    extern __shared__ float smem[];          // 16 rows x LSTRIDE f32
    const int LSTRIDE = SS + 4;              // float4-aligned rows, bank-split halves

    const int lane = threadIdx.x;            // 0..31
    const int hi   = lane >> 4;              // half-wave select
    const int ln   = lane & 15;

    const int wg   = blockIdx.x;             // B*H*(S/16) = 2048 blocks
    const int bh   = wg >> 6;                // (b*H + h)
    const int trow = wg & 63;
    const int b    = bh >> 3;
    const int h    = bh & 7;
    const int s0   = trow << 4;

    const _Float16* qbase = qws + (((size_t)bh) * SS + s0) * DH;
    const _Float16* kbase = kws + ((size_t)bh) * SS * DH;

    // ---- A fragments for the two K=32 steps (q rows, held for whole loop) ----
    v16h a0, a1;
    {
        const _Float16* qrow = qbase + ln * DH;
        v8h l0 = *(const v8h*)(qrow +      hi * 8);        // kk0: K hi*8..hi*8+7
        v8h h0 = *(const v8h*)(qrow + 16 + hi * 8);        // kk0: K 16+hi*8..
        v8h l1 = *(const v8h*)(qrow + 32 + hi * 8);        // kk1
        v8h h1 = *(const v8h*)(qrow + 48 + hi * 8);
#pragma unroll
        for (int i = 0; i < 8; ++i) {
            a0[i] = l0[i]; a0[8 + i] = h0[i];
            a1[i] = l1[i]; a1[8 + i] = h1[i];
        }
    }

    float psum[8];
#pragma unroll
    for (int i = 0; i < 8; ++i) psum[i] = 0.0f;

#pragma unroll 4
    for (int t = 0; t < SS / 16; ++t) {
        // B fragment: lane ln holds key-row (t*16+ln), K {kk*32+hi*8..+7} U {kk*32+16+hi*8..+7}
        const _Float16* krow = kbase + ((size_t)(t * 16 + ln)) * DH;
        v16h b0, b1;
        {
            v8h bl0 = *(const v8h*)(krow +      hi * 8);
            v8h bh0 = *(const v8h*)(krow + 16 + hi * 8);
            v8h bl1 = *(const v8h*)(krow + 32 + hi * 8);
            v8h bh1 = *(const v8h*)(krow + 48 + hi * 8);
#pragma unroll
            for (int i = 0; i < 8; ++i) {
                b0[i] = bl0[i]; b0[8 + i] = bh0[i];
                b1[i] = bl1[i]; b1[8 + i] = bh1[i];
            }
        }

        v8f c = {0.f, 0.f, 0.f, 0.f, 0.f, 0.f, 0.f, 0.f};
        c = __builtin_amdgcn_wmma_f32_16x16x32_f16(false, a0, false, b0,
                                                   (short)0, c, false, false);
        c = __builtin_amdgcn_wmma_f32_16x16x32_f16(false, a1, false, b1,
                                                   (short)0, c, false, false);

        // C layout: VGPR i -> row (i + hi*8), col (t*16 + ln)
#pragma unroll
        for (int i = 0; i < 8; ++i) {
            float ev = __builtin_amdgcn_exp2f(c[i]);   // bare v_exp_f32
            psum[i] += ev;
            smem[(i + hi * 8) * LSTRIDE + (t * 16 + ln)] = ev;
        }
    }

    // ---- row sums: reduce across the 16 lanes of each half-wave ----
#pragma unroll
    for (int i = 0; i < 8; ++i) {
        float v = psum[i];
        v += __shfl_xor(v, 1, 32);
        v += __shfl_xor(v, 2, 32);
        v += __shfl_xor(v, 4, 32);
        v += __shfl_xor(v, 8, 32);
        psum[i] = 1.0f / v;                  // exp2() > 0, never zero
    }

    __syncthreads();                         // order LDS writes vs cross-lane reads

    // ---- normalized store: half-wave writes its own 8 rows, float4 x 16 lanes ----
#pragma unroll
    for (int i = 0; i < 8; ++i) {
        const int   row  = i + hi * 8;
        const float isum = psum[i];
        float* orow = out + (((size_t)(b * SS + s0 + row)) * HH + h) * SS;
        const float* srow = smem + row * LSTRIDE;
#pragma unroll 4
        for (int j = 0; j < 16; ++j) {
            int c4 = (j * 16 + ln) * 4;
            float4 v = *(const float4*)(srow + c4);
            v.x *= isum; v.y *= isum; v.z *= isum; v.w *= isum;
            *(float4*)(orow + c4) = v;
        }
    }
}

// ---------------------------------------------------------------------------
extern "C" void kernel_launch(void* const* d_in, const int* in_sizes, int n_in,
                              void* d_out, int out_size, void* d_ws, size_t ws_size,
                              hipStream_t stream) {
    const float* x  = (const float*)d_in[0];   // [4,1024,512]
    const float* Wq = (const float*)d_in[1];   // [8,64,64]
    const float* bq = (const float*)d_in[2];   // [8,64]
    const float* Wk = (const float*)d_in[3];
    const float* bk = (const float*)d_in[4];
    float* out = (float*)d_out;                // [4,1024,8192]

    _Float16* qws = (_Float16*)d_ws;                       // [B,H,S,64] f16 (4 MB)
    _Float16* kws = qws + (size_t)BB * HH * SS * DH;       // +4 MB

    const int total = 2 * BB * HH * SS * DH;               // 4,194,304
    msa_proj_kernel<<<total / 256, 256, 0, stream>>>(x, Wq, bq, Wk, bk, qws, kws);

    const int nblocks  = BB * HH * (SS / 16);              // 2048
    const size_t ldsSz = (size_t)16 * (SS + 4) * sizeof(float);  // 65,792 B
    msa_attn_kernel<<<nblocks, 32, ldsSz, stream>>>(qws, kws, out);
}
